// KaomojiTagger_64699387347567
// MI455X (gfx1250) — compile-verified, matrix operand order checked by reading, not run on hardware
//
#include <hip/hip_runtime.h>
#include <hip/hip_bf16.h>
#include <math.h>

// ---- dims (match reference) ----
#define B_     64
#define T_     256
#define H_     512
#define G4H    2048      // 4*H
#define NL     34        // N_LABELS
#define START_ 32
#define STOP_  33
#define TB     16384     // T_*B_

typedef __attribute__((ext_vector_type(16))) _Float16 v16h;
typedef __attribute__((ext_vector_type(8)))  _Float16 v8h;
typedef __attribute__((ext_vector_type(8)))  float    v8f;

__device__ __forceinline__ float kt_sig(float x) { return 1.f / (1.f + expf(-x)); }

// ------------------------------------------------------------------
// CDNA5 async global->LDS staging (GLOBAL_LOAD_ASYNC_TO_LDS_B128,
// tracked by ASYNCcnt). Builtin signature on this toolchain (from the
// round-2 diagnostic): arg0 = v4i in AS1 (global), arg1 = LDS side,
// then two immediates (offset, cpol).
// ------------------------------------------------------------------
#if defined(__HIP_DEVICE_COMPILE__) && __has_builtin(__builtin_amdgcn_global_load_async_to_lds_b128)
#define KT_HAS_ASYNC_LDS 1
#endif

#if defined(KT_HAS_ASYNC_LDS)
typedef int kt_v4i __attribute__((vector_size(16)));
typedef kt_v4i __attribute__((address_space(1))) kt_v4i_glob;
typedef kt_v4i __attribute__((address_space(3))) kt_v4i_lds;

__device__ __forceinline__ void kt_async_cp16(const _Float16* g, _Float16* l) {
  __builtin_amdgcn_global_load_async_to_lds_b128(
      (kt_v4i_glob*)g, (kt_v4i_lds*)l, 0, 0);
}
__device__ __forceinline__ void kt_wait_async() {
#if __has_builtin(__builtin_amdgcn_s_wait_asynccnt)
  __builtin_amdgcn_s_wait_asynccnt(0);
#else
  asm volatile("s_wait_asynccnt 0x0" ::: "memory");
#endif
}
#endif

// ------------------------------------------------------------------
// WMMA fragment loaders. ISA 7.12.2 (16-bit A 16x32 layout):
//  lane<16 : row M=lane,    halves 0..7 -> K=k0+0..7,  halves 8..15 -> K=k0+16..23
//  lane>=16: row M=lane-16, halves 0..7 -> K=k0+8..15, halves 8..15 -> K=k0+24..31
// For B (K x 16) we exploit that weights are stored [N,K] row-major, so the
// B-operand column n is a contiguous K-run: identical load pattern with n as row.
// ------------------------------------------------------------------
__device__ __forceinline__ v16h kt_load_frag(const _Float16* __restrict__ base,
                                             int ld, int row0, int k0) {
  const int lane = threadIdx.x & 31;
  const _Float16* p = base + (size_t)(row0 + (lane & 15)) * ld + k0 + ((lane >> 4) << 3);
  union { v16h v; struct { v8h lo, hi; } s; } u;
  u.s.lo = *(const v8h*)(p);        // 16B
  u.s.hi = *(const v8h*)(p + 16);   // 16B, K offset +16 halves
  return u.v;
}

// C/D 16x16 f32 layout: lane<16 -> N=lane, M=vgpr r; lane>=16 -> N=lane-16, M=8+r.
__device__ __forceinline__ void kt_store_tile_bias(float* __restrict__ C, int ldc,
                                                   int m0, int n0, v8f acc,
                                                   const float* __restrict__ bias) {
  const int lane = threadIdx.x & 31;
  const int col  = n0 + (lane & 15);
  const int rb   = m0 + ((lane >> 4) << 3);
  const float bv = bias[col];
#pragma unroll
  for (int r = 0; r < 8; ++r) C[(size_t)(rb + r) * ldc + col] = acc[r] + bv;
}

// ------------------------------------------------------------------
// fp32 -> fp16 weight/activation conversion
// ------------------------------------------------------------------
__global__ void kt_f32_to_f16(const float* __restrict__ s, _Float16* __restrict__ d, int n) {
  int i = blockIdx.x * blockDim.x + threadIdx.x;
  if (i < n) d[i] = (_Float16)s[i];
}

// ------------------------------------------------------------------
// Embedding gather: x0[t*B+b, 0:448] = concat(char_embed[src], feat0..2)
// ------------------------------------------------------------------
__global__ void kt_embed(const int* __restrict__ src, const int* __restrict__ f0,
                         const int* __restrict__ f1, const int* __restrict__ f2,
                         const float* __restrict__ cemb, const float* __restrict__ femb,
                         _Float16* __restrict__ x) {
  const int r = blockIdx.x;            // r = t*B + b
  const int t = r >> 6, b = r & 63;
  const int s  = src[b * T_ + t];
  const int v0 = f0[b * T_ + t];
  const int v1 = f1[b * T_ + t];
  const int v2 = f2[b * T_ + t];
  _Float16* row = x + (size_t)r * 448;
  for (int c = threadIdx.x; c < 448; c += blockDim.x) {
    float val;
    if (c < 256) {
      val = cemb[(size_t)s * 256 + c];
    } else {
      const int fi = (c - 256) >> 6, fc = (c - 256) & 63;
      const int v = (fi == 0) ? v0 : ((fi == 1) ? v1 : v2);
      val = femb[(size_t)(fi * 64 + v) * 64 + fc];
    }
    row[c] = (_Float16)val;
  }
}

// ------------------------------------------------------------------
// Big input GEMM: Gpre[M=16384, N=2048] = A[M,K] @ W[N,K]^T + bias[N]
// 256 threads = 8 waves as 4(M)x2(N); wave tile 32x32 (2x2 WMMA tiles).
// Block tile 128x64. grid = (2048/64, 16384/128) = (32,128).
// ------------------------------------------------------------------
__global__ __launch_bounds__(256) void kt_gemm_wih(const _Float16* __restrict__ A,
                                                   const _Float16* __restrict__ W,
                                                   const float* __restrict__ bias,
                                                   float* __restrict__ C, int K) {
  const int wave = threadIdx.x >> 5;
  const int lane = threadIdx.x & 31;
  const int m0 = blockIdx.y * 128 + (wave >> 1) * 32;
  const int n0 = blockIdx.x * 64 + (wave & 1) * 32;

  v8f zero = {0.f, 0.f, 0.f, 0.f, 0.f, 0.f, 0.f, 0.f};
  v8f acc00 = zero, acc01 = zero, acc10 = zero, acc11 = zero;

  for (int k0 = 0; k0 < K; k0 += 32) {
    if (k0 + 64 < K) {  // prefetch ahead -> global_prefetch_b8
      __builtin_prefetch(A + (size_t)(m0 + (lane & 15)) * K + k0 + 64, 0, 0);
      __builtin_prefetch(W + (size_t)(n0 + (lane & 15)) * K + k0 + 64, 0, 0);
    }
    v16h a0 = kt_load_frag(A, K, m0,      k0);
    v16h a1 = kt_load_frag(A, K, m0 + 16, k0);
    v16h b0 = kt_load_frag(W, K, n0,      k0);
    v16h b1 = kt_load_frag(W, K, n0 + 16, k0);
    acc00 = __builtin_amdgcn_wmma_f32_16x16x32_f16(false, a0, false, b0, (short)0, acc00, false, false);
    acc01 = __builtin_amdgcn_wmma_f32_16x16x32_f16(false, a0, false, b1, (short)0, acc01, false, false);
    acc10 = __builtin_amdgcn_wmma_f32_16x16x32_f16(false, a1, false, b0, (short)0, acc10, false, false);
    acc11 = __builtin_amdgcn_wmma_f32_16x16x32_f16(false, a1, false, b1, (short)0, acc11, false, false);
  }
  kt_store_tile_bias(C, G4H, m0,      n0,      acc00, bias);
  kt_store_tile_bias(C, G4H, m0,      n0 + 16, acc01, bias);
  kt_store_tile_bias(C, G4H, m0 + 16, n0,      acc10, bias);
  kt_store_tile_bias(C, G4H, m0 + 16, n0 + 16, acc11, bias);
}

// ------------------------------------------------------------------
// Grid-wide arrive/spin barrier for the persistent recurrent kernel.
// Counter is monotonically increasing; target = 64 * phase_index.
// ------------------------------------------------------------------
__device__ __forceinline__ void kt_gsync(unsigned long long* c, unsigned long long target) {
  __syncthreads();
  if (threadIdx.x == 0) {
    __threadfence();
    atomicAdd(c, 1ull);
    volatile unsigned long long* vc = (volatile unsigned long long*)c;
    while (*vc < target) { __builtin_amdgcn_s_sleep(2); }
    __threadfence();
  }
  __syncthreads();
}

// ------------------------------------------------------------------
// Persistent LSTM recurrence. 64 WGs x 8 waves = 512 waves = one 16x16 tile
// each of the 64x2048 gate matrix. Per step:
//   stage  : h16 (64KB) -> LDS via async global->LDS copies (ASYNCcnt path)
//   phase A: gbuf = Gpre[t] + h_lds @ Whh16^T   (WMMA, A-frags from LDS)
//   phase B: LSTM cell elementwise, masked update of h/c, write y (f16)
// Whh16 (2MB) streams from L2 once per step across the grid; h comes from
// LDS, keeping L2 latency off the serial critical path.
// ------------------------------------------------------------------
__global__ __launch_bounds__(256) void kt_lstm_recur(const float* __restrict__ gpre,
                                                     const _Float16* __restrict__ U,
                                                     float* __restrict__ gbuf,
                                                     float* __restrict__ hbuf,
                                                     float* __restrict__ cbuf,
                                                     _Float16* __restrict__ h16,
                                                     _Float16* __restrict__ xout,
                                                     const int* __restrict__ length,
                                                     int dir, unsigned long long* sync) {
  __shared__ _Float16 hs[B_ * H_];   // 64 KB of the 320 KB WGP LDS
  const int wg   = blockIdx.x * 8 + (threadIdx.x >> 5);
  const int m0   = (wg & 3) << 4;    // batch-tile origin (0..48)
  const int n0   = (wg >> 2) << 4;   // gate-column tile origin (0..2032)
  const int lane = threadIdx.x & 31;
  const int tid  = blockIdx.x * 256 + threadIdx.x;   // 0..16383
  unsigned long long phase = 0;
  const v8f zero = {0.f, 0.f, 0.f, 0.f, 0.f, 0.f, 0.f, 0.f};

  for (int s = 0; s < T_; ++s) {
    const int t = dir ? (T_ - 1 - s) : s;

    // ---- stage h into LDS (16B per thread per iter, 16 iters) ----
#if defined(KT_HAS_ASYNC_LDS)
#pragma unroll
    for (int i = threadIdx.x * 8; i < B_ * H_; i += 2048)
      kt_async_cp16(h16 + i, &hs[i]);
    kt_wait_async();
#else
#pragma unroll
    for (int i = threadIdx.x * 8; i < B_ * H_; i += 2048)
      *(v8h*)&hs[i] = *(const v8h*)(h16 + i);
#endif
    __syncthreads();

    // ---- phase A: gate preactivations ----
    v8f acc = zero;
    for (int k0 = 0; k0 < H_; k0 += 32) {
      v16h a = kt_load_frag(hs, H_, m0, k0);   // ds_load_b128 path
      v16h b = kt_load_frag(U,  H_, n0, k0);   // L2 stream
      acc = __builtin_amdgcn_wmma_f32_16x16x32_f16(false, a, false, b, (short)0, acc, false, false);
    }
    {
      const int col = n0 + (lane & 15);
      const int rb  = m0 + ((lane >> 4) << 3);
      const float* gp = gpre + (size_t)t * B_ * G4H;
#pragma unroll
      for (int r = 0; r < 8; ++r)
        gbuf[(size_t)(rb + r) * G4H + col] = acc[r] + gp[(size_t)(rb + r) * G4H + col];
    }
    kt_gsync(sync, (++phase) * 64ull);

    // ---- phase B: LSTM cell (2 elements per thread: 64*512 / 16384) ----
    for (int e = tid; e < B_ * H_; e += 16384) {
      const int b = e >> 9, j = e & 511;
      const float gi = gbuf[(size_t)b * G4H + j];
      const float gf = gbuf[(size_t)b * G4H + 512 + j];
      const float gg = gbuf[(size_t)b * G4H + 1024 + j];
      const float go = gbuf[(size_t)b * G4H + 1536 + j];
      const float cprev = cbuf[e];
      const float hprev = hbuf[e];
      const float cn = kt_sig(gf) * cprev + kt_sig(gi) * tanhf(gg);
      const float hn = kt_sig(go) * tanhf(cn);
      const bool  m  = (t < length[b]);
      const float hs2 = m ? hn : hprev;
      const float cs  = m ? cn : cprev;
      hbuf[e] = hs2;
      cbuf[e] = cs;
      h16[e]  = (_Float16)hs2;
      xout[(size_t)(t * B_ + b) * 1024 + dir * 512 + j] = (_Float16)(m ? hn : 0.f);
    }
    kt_gsync(sync, (++phase) * 64ull);
  }
}

// ------------------------------------------------------------------
// logits[b,t,l] = x2[t*B+b, :] . Wcls[l, :] + bcls[l]   (tiny, 1.1 GF)
// ------------------------------------------------------------------
__global__ void kt_logits(const _Float16* __restrict__ X, const float* __restrict__ W,
                          const float* __restrict__ bias, float* __restrict__ out) {
  const int idx = blockIdx.x * blockDim.x + threadIdx.x;
  if (idx >= TB * NL) return;
  const int r = idx / NL, l = idx - r * NL;
  const int t = r >> 6, b = r & 63;
  const _Float16* x = X + (size_t)r * 1024;
  const float*    w = W + (size_t)l * 1024;
  float s = bias[l];
  for (int k = 0; k < 1024; ++k) s += (float)x[k] * w[k];
  out[((size_t)b * T_ + t) * NL + l] = s;
}

// ------------------------------------------------------------------
// CRF: one block per batch. Forward algorithm (alpha recursion) with
// transitions cached in LDS, plus emit and transition path scores.
// out[b] = -(seq_score - norm)
// ------------------------------------------------------------------
__global__ __launch_bounds__(64) void kt_crf(const float* __restrict__ logits,
                                             const int* __restrict__ trg,
                                             const int* __restrict__ length,
                                             const float* __restrict__ trans,
                                             float* __restrict__ out) {
  __shared__ float tr[NL * NL];
  __shared__ float alpha[NL];
  __shared__ float red[64];
  const int b = blockIdx.x;
  const int j = threadIdx.x;
  const int len = length[b];

  for (int i = j; i < NL * NL; i += 64) tr[i] = trans[i];
  if (j < NL) alpha[j] = (j == START_) ? 0.f : -10000.f;
  __syncthreads();

  const float* lg = logits + (size_t)b * T_ * NL;
  for (int t = 0; t < T_; ++t) {
    float an = 0.f;
    if (j < NL) {
      float mx = -1e30f;
      for (int k = 0; k < NL; ++k) mx = fmaxf(mx, tr[j * NL + k] + alpha[k]);
      float sm = 0.f;
      for (int k = 0; k < NL; ++k) sm += expf(tr[j * NL + k] + alpha[k] - mx);
      an = lg[t * NL + j] + mx + logf(sm);
    }
    __syncthreads();
    if (j < NL && t < len) alpha[j] = an;
    __syncthreads();
  }
  if (j < NL) alpha[j] += tr[STOP_ * NL + j];
  __syncthreads();

  // emit score
  float acc = 0.f;
  for (int t = j; t < T_; t += 64)
    if (t < len) acc += lg[t * NL + trg[b * T_ + t]];
  red[j] = acc;
  __syncthreads();
  for (int s2 = 32; s2 > 0; s2 >>= 1) { if (j < s2) red[j] += red[j + s2]; __syncthreads(); }
  const float emit = red[0];
  __syncthreads();

  // transition score: sum_{i=0..len} trans[le[i+1], le[i]]
  float ta = 0.f;
  for (int i = j; i <= T_; i += 64) {
    if (i <= len) {
      const int prev = (i == 0) ? START_ : trg[b * T_ + i - 1];
      const int nxt  = (i < len) ? trg[b * T_ + i] : STOP_;
      ta += tr[nxt * NL + prev];
    }
  }
  red[j] = ta;
  __syncthreads();
  for (int s2 = 32; s2 > 0; s2 >>= 1) { if (j < s2) red[j] += red[j + s2]; __syncthreads(); }
  const float trsc = red[0];

  if (j == 0) {
    float mx = -1e30f;
    for (int k = 0; k < NL; ++k) mx = fmaxf(mx, alpha[k]);
    float sm = 0.f;
    for (int k = 0; k < NL; ++k) sm += expf(alpha[k] - mx);
    const float norm = mx + logf(sm);
    out[b] = -((trsc + emit) - norm);
  }
}

// ------------------------------------------------------------------
// Host orchestration.
// Input order (jax pytree flatten, dicts by sorted key):
//  0 src, 1 trg, 2 feat0, 3 feat1, 4 feat2, 5 length,
//  6 W (34x1024), 7 b (34), 8 char_embed (8000x256), 9 feat_embed (3x64x64),
//  10.. lstm[layer][dir] as (Whh, Wih, b) x4,  22 transitions (34x34)
// ------------------------------------------------------------------
extern "C" void kernel_launch(void* const* d_in, const int* in_sizes, int n_in,
                              void* d_out, int out_size, void* d_ws, size_t ws_size,
                              hipStream_t stream) {
  (void)in_sizes; (void)n_in; (void)out_size; (void)ws_size;

  const int*   src    = (const int*)d_in[0];
  const int*   trg    = (const int*)d_in[1];
  const int*   f0     = (const int*)d_in[2];
  const int*   f1     = (const int*)d_in[3];
  const int*   f2     = (const int*)d_in[4];
  const int*   length = (const int*)d_in[5];
  const float* Wcls   = (const float*)d_in[6];
  const float* bcls   = (const float*)d_in[7];
  const float* cemb   = (const float*)d_in[8];
  const float* femb   = (const float*)d_in[9];
  const float* trans  = (const float*)d_in[22];
  float*       out    = (float*)d_out;

  char* ws = (char*)d_ws;
  size_t off = 0;
  auto alloc = [&](size_t bytes) -> char* {
    off = (off + 255) & ~(size_t)255;
    char* p = ws + off;
    off += bytes;
    return p;
  };

  _Float16* x0   = (_Float16*)alloc((size_t)TB * 448 * 2);   // layer1 input (f16)
  _Float16* x1   = (_Float16*)alloc((size_t)TB * 1024 * 2);  // layer1 output / layer2 input
  _Float16* x2   = (_Float16*)alloc((size_t)TB * 1024 * 2);  // layer2 output
  float*    gpre = (float*)alloc((size_t)TB * G4H * 4);      // gate preactivations (134MB)
  _Float16* w16  = (_Float16*)alloc((size_t)G4H * 1024 * 2); // Wih f16
  _Float16* u16  = (_Float16*)alloc((size_t)G4H * H_ * 2);   // Whh f16
  float*    gbuf = (float*)alloc((size_t)B_ * G4H * 4);      // per-step gates
  float*    hbuf = (float*)alloc((size_t)B_ * H_ * 4);
  float*    cbuf = (float*)alloc((size_t)B_ * H_ * 4);
  _Float16* h16  = (_Float16*)alloc((size_t)B_ * H_ * 2);
  float*    lgts = (float*)alloc((size_t)B_ * T_ * NL * 4);
  unsigned long long* sync = (unsigned long long*)alloc(256);

  kt_embed<<<TB, 256, 0, stream>>>(src, f0, f1, f2, cemb, femb, x0);

  for (int layer = 0; layer < 2; ++layer) {
    const int K = layer ? 1024 : 448;
    const _Float16* xin = layer ? x1 : x0;
    _Float16* xout = layer ? x2 : x1;
    for (int dir = 0; dir < 2; ++dir) {
      const int base = 10 + (layer * 2 + dir) * 3;
      const float* Whh = (const float*)d_in[base + 0];
      const float* Wih = (const float*)d_in[base + 1];
      const float* bg  = (const float*)d_in[base + 2];
      const int nW = G4H * K, nU = G4H * H_;

      kt_f32_to_f16<<<(nW + 255) / 256, 256, 0, stream>>>(Wih, w16, nW);
      kt_f32_to_f16<<<(nU + 255) / 256, 256, 0, stream>>>(Whh, u16, nU);

      kt_gemm_wih<<<dim3(32, 128), 256, 0, stream>>>(xin, w16, bg, gpre, K);

      (void)hipMemsetAsync(hbuf, 0, (size_t)B_ * H_ * 4, stream);
      (void)hipMemsetAsync(cbuf, 0, (size_t)B_ * H_ * 4, stream);
      (void)hipMemsetAsync(h16,  0, (size_t)B_ * H_ * 2, stream);
      (void)hipMemsetAsync(sync, 0, 8, stream);

      kt_lstm_recur<<<64, 256, 0, stream>>>(gpre, u16, gbuf, hbuf, cbuf, h16,
                                            xout, length, dir, sync);
    }
  }

  kt_logits<<<(TB * NL + 255) / 256, 256, 0, stream>>>(x2, Wcls, bcls, lgts);
  kt_crf<<<B_, 64, 0, stream>>>(lgts, trg, length, trans, out);
}
